// MoETransformerEncoderLayer_52750788329547
// MI455X (gfx1250) — matrix operand order repaired and checked by reading, not compiled, not run on hardware
//
// MoE Transformer encoder layer for MI455X (gfx1250, wave32, WMMA + TDM).
// Heavy GEMMs + attention use v_wmma_f32_16x16x32_bf16 (bf16 in, fp32 acc).
// GEMM tiles are fed by the Tensor Data Mover (tensor_load_to_lds, TENSORcnt)
// with KT=64 ping-pong double buffering; b128 staging is the compile fallback.
// Roofline: ~1.7 TFLOP vs ~0.6 GB weight traffic -> compute bound -> WMMA path.
#include <hip/hip_runtime.h>
#include <stdint.h>

#define T_TOK 2048
#define MDIM 4096
#define DHDIM 4096
#define KVDIM 1024
#define NEXP 8
#define NHEAD 16
#define HDIM 256
#define SCALE_ATT 0.0625f   // HD^-0.5

typedef __bf16 bf16_t;
typedef __attribute__((ext_vector_type(16))) __bf16 bf16x16;
typedef __attribute__((ext_vector_type(8)))  __bf16 bf16x8;
typedef __attribute__((ext_vector_type(8)))  float  f32x8;
typedef __attribute__((ext_vector_type(4)))  unsigned int u32x4;
typedef __attribute__((ext_vector_type(4)))  int i32x4;
typedef __attribute__((ext_vector_type(8)))  int i32x8;

#if __has_builtin(__builtin_amdgcn_tensor_load_to_lds)
#define USE_TDM 1
#else
#define USE_TDM 0
#endif

union BF16Frag { bf16x16 v; bf16x8 h[2]; };

static __device__ __forceinline__ bf16_t f2bf(float f) {
  union { float f; uint32_t u; } in; in.f = f;
  uint32_t r = in.u + 0x7fffu + ((in.u >> 16) & 1u);   // round-to-nearest-even
  union { uint16_t u; bf16_t b; } out; out.u = (uint16_t)(r >> 16);
  return out.b;
}
static __device__ __forceinline__ f32x8 zero8() {
  f32x8 z;
#pragma unroll
  for (int i = 0; i < 8; ++i) z[i] = 0.f;
  return z;
}
static __device__ __forceinline__ f32x8 wmma_bf16(const BF16Frag& a, const BF16Frag& b, f32x8 c) {
  return __builtin_amdgcn_wmma_f32_16x16x32_bf16(false, a.v, false, b.v, (short)0, c, false, false);
}

#if USE_TDM
// Issue one TDM 2D tile load: tile_rows x tile_cols bf16 elements, row stride
// row_stride (elements), packed contiguously into LDS at byte offset lds_off.
// D# per CDNA5 ISA ch.8: group0 {count=1, lds_addr, global_addr[56:0], type=2},
// group1 {data_size=2B, huge tensor dims (no OOB clip), tile dims, dim0 stride}.
static __device__ __forceinline__ void tdm_load_2d(
    uint32_t lds_off, const void* gptr, uint32_t tile_cols,
    uint32_t tile_rows, uint32_t row_stride) {
  const uint64_t ga = (uint64_t)(uintptr_t)gptr;
  u32x4 g0; i32x8 g1; i32x4 g2; i32x4 g3;
  g0[0] = 1u;                                   // count=1, user descriptor
  g0[1] = lds_off;                              // lds_addr (bytes)
  g0[2] = (uint32_t)ga;                         // global_addr[31:0]
  g0[3] = ((uint32_t)(ga >> 32) & 0x01FFFFFFu) | 0x80000000u;  // addr[56:32] | type=2
  g1[0] = 1 << 16;                              // data_size = 1 -> 2 bytes
  g1[1] = 0;                                    // tensor_dim0 = 1<<24 (low 16 = 0)
  g1[2] = 0x0100;                               // tensor_dim0 hi | tensor_dim1 lo
  g1[3] = (int)(0x0100u | (tile_cols << 16));   // tensor_dim1 hi | tile_dim0
  g1[4] = (int)tile_rows;                       // tile_dim1 (tile_dim2 = 0)
  g1[5] = (int)row_stride;                      // tensor_dim0_stride (elements)
  g1[6] = 0; g1[7] = 0;
  g2[0] = g2[1] = g2[2] = g2[3] = 0;
  g3[0] = g3[1] = g3[2] = g3[3] = 0;
#if __clang_major__ >= 23
  i32x8 g4;
#pragma unroll
  for (int i = 0; i < 8; ++i) g4[i] = 0;
  __builtin_amdgcn_tensor_load_to_lds(g0, g1, g2, g3, g4, 0);
#else
  __builtin_amdgcn_tensor_load_to_lds(g0, g1, g2, g3, 0);
#endif
}
#endif

static __device__ __forceinline__ void wait_tensor() {
#if USE_TDM
#if __has_builtin(__builtin_amdgcn_s_wait_tensorcnt)
  __builtin_amdgcn_s_wait_tensorcnt(0);
#else
  asm volatile("s_wait_tensorcnt 0x0" ::: "memory");
#endif
#endif
}

// ---------------------------------------------------------------- transpose+convert
// W (K x N, f32, row-major) -> Wt (N x K, bf16, row-major)
__global__ void __launch_bounds__(256) transpose_convert_kernel(
    const float* __restrict__ W, bf16_t* __restrict__ Wt, int K, int N) {
  __shared__ float tile[32][33];
  const int n0 = blockIdx.x * 32, k0 = blockIdx.y * 32;
  const int tx = threadIdx.x & 31, ty = threadIdx.x >> 5;
#pragma unroll
  for (int i = 0; i < 32; i += 8)
    tile[ty + i][tx] = W[(size_t)(k0 + ty + i) * N + n0 + tx];
  __syncthreads();
#pragma unroll
  for (int i = 0; i < 32; i += 8)
    Wt[(size_t)(n0 + ty + i) * K + k0 + tx] = f2bf(tile[tx][ty + i]);
}

// ---------------------------------------------------------------- layernorm (width 4096)
__global__ void __launch_bounds__(256) layernorm_kernel(
    const float* __restrict__ x, const float* __restrict__ g, const float* __restrict__ b,
    bf16_t* __restrict__ out_bf, float* __restrict__ out_f) {
  const int row = blockIdx.x;
  const float* xr = x + (size_t)row * MDIM;
  float vals[16];
  float s = 0.f;
#pragma unroll
  for (int i = 0; i < 16; ++i) { vals[i] = xr[threadIdx.x + i * 256]; s += vals[i]; }
  __shared__ float red[256];
  red[threadIdx.x] = s; __syncthreads();
  for (int off = 128; off > 0; off >>= 1) {
    if (threadIdx.x < off) red[threadIdx.x] += red[threadIdx.x + off];
    __syncthreads();
  }
  const float mu = red[0] * (1.f / MDIM);
  __syncthreads();
  float s2 = 0.f;
#pragma unroll
  for (int i = 0; i < 16; ++i) { float d = vals[i] - mu; s2 += d * d; }
  red[threadIdx.x] = s2; __syncthreads();
  for (int off = 128; off > 0; off >>= 1) {
    if (threadIdx.x < off) red[threadIdx.x] += red[threadIdx.x + off];
    __syncthreads();
  }
  const float rstd = rsqrtf(red[0] * (1.f / MDIM) + 1e-5f);
#pragma unroll
  for (int i = 0; i < 16; ++i) {
    int c = threadIdx.x + i * 256;
    float y = (vals[i] - mu) * rstd * g[c] + b[c];
    out_bf[(size_t)row * MDIM + c] = f2bf(y);
    if (out_f) out_f[(size_t)row * MDIM + c] = y;
  }
}

// ---------------------------------------------------------------- WMMA GEMM (TDM-fed)
// C(M_ x N) = A(M_ x K, bf16) * Bt(N x K, bf16)^T, templated epilogue.
// MODE 0: outF = v + bias
// MODE 1: outB = bf16(v + bias)
// MODE 2: outF = resid + v + bias
// MODE 3: acc += gates[t,e] * silu(v + bias) * xV    (fused MoE expert)
template <int MODE>
__global__ void __launch_bounds__(256) gemm_kernel(
    const bf16_t* __restrict__ A, int lda,
    const bf16_t* __restrict__ Bt, int ldb,
    const float* __restrict__ bias, int K, int ldo,
    float* __restrict__ outF, bf16_t* __restrict__ outB,
    const float* __restrict__ resid,
    const float* __restrict__ gates, int expert,
    const float* __restrict__ xV, float* __restrict__ acc) {
  __shared__ __align__(16) bf16_t lA[2][128 * 64];   // ping-pong 16KB each
  __shared__ __align__(16) bf16_t lB[2][128 * 64];
  const int tid  = threadIdx.x;
  const int lane = tid & 31;
  const int wave = tid >> 5;
  const int wm   = (wave >> 1) * 32;
  const int wn   = (wave & 1) * 64;
  const int half = lane >> 4;
  const int l16  = lane & 15;
  const size_t m0 = (size_t)blockIdx.y * 128;
  const size_t n0 = (size_t)blockIdx.x * 128;
  const size_t baseA = m0 * (size_t)lda;
  const size_t baseB = n0 * (size_t)ldb;

  f32x8 c[2][4];
#pragma unroll
  for (int i = 0; i < 2; ++i)
#pragma unroll
    for (int j = 0; j < 4; ++j) c[i][j] = zero8();

#if USE_TDM
  const uint32_t offA[2] = { (uint32_t)(uintptr_t)&lA[0][0], (uint32_t)(uintptr_t)&lA[1][0] };
  const uint32_t offB[2] = { (uint32_t)(uintptr_t)&lB[0][0], (uint32_t)(uintptr_t)&lB[1][0] };
#endif

  // stage k-tile [kt, kt+64) into buffer `buf`
  auto stage = [&](int buf, int kt) {
#if USE_TDM
    if (tid == 0) {  // one TDM descriptor per matrix tile; EXEC!=0 lane0 issues
      tdm_load_2d(offA[buf], A  + baseA + kt, 64, 128, (uint32_t)lda);
      tdm_load_2d(offB[buf], Bt + baseB + kt, 64, 128, (uint32_t)ldb);
    }
#else
#pragma unroll
    for (int i = 0; i < 4; ++i) {
      int lin = tid + i * 256;   // 1024 x 16B per matrix
      int row = lin >> 3;
      int c8  = lin & 7;
      *(uint4*)&lA[buf][row * 64 + c8 * 8] = *(const uint4*)&A [baseA + (size_t)row * lda + kt + c8 * 8];
      *(uint4*)&lB[buf][row * 64 + c8 * 8] = *(const uint4*)&Bt[baseB + (size_t)row * ldb + kt + c8 * 8];
    }
#endif
  };

  stage(0, 0);
  wait_tensor();
  __syncthreads();

  for (int kt = 0; kt < K; kt += 64) {
    const int buf = (kt >> 6) & 1;
    if (kt + 64 < K) stage(buf ^ 1, kt + 64);   // DMA next tile under compute
    const bf16_t* As = lA[buf];
    const bf16_t* Bs = lB[buf];
#pragma unroll
    for (int kc = 0; kc < 2; ++kc) {
      BF16Frag af[2], bfr[4];
#pragma unroll
      for (int ti = 0; ti < 2; ++ti) {
        int r = wm + ti * 16 + l16;
        af[ti].h[0] = *(const bf16x8*)&As[r * 64 + kc * 32 + half * 8];
        af[ti].h[1] = *(const bf16x8*)&As[r * 64 + kc * 32 + 16 + half * 8];
      }
#pragma unroll
      for (int tj = 0; tj < 4; ++tj) {
        int n = wn + tj * 16 + l16;
        bfr[tj].h[0] = *(const bf16x8*)&Bs[n * 64 + kc * 32 + half * 8];
        bfr[tj].h[1] = *(const bf16x8*)&Bs[n * 64 + kc * 32 + 16 + half * 8];
      }
#pragma unroll
      for (int ti = 0; ti < 2; ++ti)
#pragma unroll
        for (int tj = 0; tj < 4; ++tj)
          c[ti][tj] = wmma_bf16(af[ti], bfr[tj], c[ti][tj]);
    }
    wait_tensor();        // next tile landed (issuing wave; others pass)
    __syncthreads();      // publish to all waves / allow overwrite
  }

#pragma unroll
  for (int ti = 0; ti < 2; ++ti)
#pragma unroll
    for (int tj = 0; tj < 4; ++tj)
#pragma unroll
      for (int j = 0; j < 8; ++j) {
        size_t row = m0 + wm + ti * 16 + j + half * 8;
        size_t col = n0 + wn + tj * 16 + l16;
        float v = c[ti][tj][j];
        if (bias) v += bias[col];
        size_t idx = row * (size_t)ldo + col;
        if (MODE == 0) {
          outF[idx] = v;
        } else if (MODE == 1) {
          outB[idx] = f2bf(v);
        } else if (MODE == 2) {
          outF[idx] = resid[idx] + v;
        } else {  // MODE 3
          float gte = gates[row * NEXP + expert];
          float sv = v / (1.f + __expf(-v));
          acc[idx] += gte * sv * xV[idx];
        }
      }
}

// ---------------------------------------------------------------- WMMA flash attention
// q: (B*N, M) bf16 with head offset h*HD; k,v: (B*N, KV) bf16 with group offset.
// GQA via jnp.tile -> head h uses KV group (h % 4).
__global__ void __launch_bounds__(128) attention_kernel(
    const bf16_t* __restrict__ q, const bf16_t* __restrict__ k,
    const bf16_t* __restrict__ v, bf16_t* __restrict__ o) {
  const int n0   = blockIdx.x * 64;
  const int h    = blockIdx.y;
  const int b    = blockIdx.z;
  const int grp  = h & 3;   // tile semantics: head -> group h % NG
  const int tid  = threadIdx.x;
  const int lane = tid & 31;
  const int wave = tid >> 5;
  const int half = lane >> 4;
  const int l16  = lane & 15;

  __shared__ __align__(32) bf16_t vt[256 * 32];        // V^T tile: [d][key]
  __shared__ __align__(16) bf16_t pP[4][16 * 32];      // per-wave P tile

  const bf16_t* qb = q + ((size_t)(b * 1024 + n0 + wave * 16)) * MDIM + h * HDIM;
  const bf16_t* kb = k + ((size_t)(b * 1024)) * KVDIM + grp * HDIM;
  const bf16_t* vb = v + ((size_t)(b * 1024)) * KVDIM + grp * HDIM;

  f32x8 accd[16];
#pragma unroll
  for (int i = 0; i < 16; ++i) accd[i] = zero8();
  float mrow[8], lrow[8];
#pragma unroll
  for (int j = 0; j < 8; ++j) { mrow[j] = -1e30f; lrow[j] = 0.f; }

  for (int ck = 0; ck < 1024; ck += 32) {
    __syncthreads();
    // stage V^T (32 keys x 256 d): coalesced 8-wide reads, scalar transposed writes
#pragma unroll
    for (int i = 0; i < 8; ++i) {
      int lin = i * 128 + tid;            // 0..1023 chunks of 8 d
      int key = lin >> 5;
      int dg  = lin & 31;
      bf16x8 chunk = *(const bf16x8*)&vb[(size_t)(ck + key) * KVDIM + dg * 8];
#pragma unroll
      for (int jj = 0; jj < 8; ++jj) vt[(dg * 8 + jj) * 32 + key] = chunk[jj];
    }
    // S = Q * K^T  (two 16x16 tiles: keys ck..+15 and ck+16..+31)
    f32x8 s0 = zero8(), s1 = zero8();
#pragma unroll
    for (int dc = 0; dc < 8; ++dc) {
      BF16Frag aq, b0, b1;
      const bf16_t* qr = qb + (size_t)l16 * MDIM + dc * 32;
      aq.h[0] = *(const bf16x8*)(qr + half * 16);
      aq.h[1] = *(const bf16x8*)(qr + half * 16 + 8);
      b0.v = *(const bf16x16*)(kb + (size_t)(ck + l16) * KVDIM + dc * 32 + half * 16);
      b1.v = *(const bf16x16*)(kb + (size_t)(ck + 16 + l16) * KVDIM + dc * 32 + half * 16);
      s0 = wmma_bf16(aq, b0, s0);
      s1 = wmma_bf16(aq, b1, s1);
    }
    // online softmax (row j+half*8 lives in reg j across 16 lanes)
    float corr[8];
#pragma unroll
    for (int j = 0; j < 8; ++j) {
      float a0 = s0[j] * SCALE_ATT, a1 = s1[j] * SCALE_ATT;
      float mx = fmaxf(a0, a1);
#pragma unroll
      for (int off = 1; off < 16; off <<= 1) mx = fmaxf(mx, __shfl_xor(mx, off, 32));
      float nm = fmaxf(mrow[j], mx);
      corr[j] = __expf(mrow[j] - nm);
      float p0 = __expf(a0 - nm), p1 = __expf(a1 - nm);
      float rs = p0 + p1;
#pragma unroll
      for (int off = 1; off < 16; off <<= 1) rs += __shfl_xor(rs, off, 32);
      lrow[j] = lrow[j] * corr[j] + rs;
      mrow[j] = nm;
      pP[wave][(j + half * 8) * 32 + l16]      = f2bf(p0);
      pP[wave][(j + half * 8) * 32 + 16 + l16] = f2bf(p1);
    }
#pragma unroll
    for (int dt = 0; dt < 16; ++dt)
#pragma unroll
      for (int j = 0; j < 8; ++j) accd[dt][j] *= corr[j];
    __syncthreads();   // vt staged + P visible
    // O += P * V
    BF16Frag ap;
    {
      const bf16_t* pr = &pP[wave][l16 * 32];
      ap.h[0] = *(const bf16x8*)(pr + half * 16);
      ap.h[1] = *(const bf16x8*)(pr + half * 16 + 8);
    }
#pragma unroll
    for (int dt = 0; dt < 16; ++dt) {
      BF16Frag bv;
      bv.v = *(const bf16x16*)&vt[(dt * 16 + l16) * 32 + half * 16];
      accd[dt] = wmma_bf16(ap, bv, accd[dt]);
    }
  }
#pragma unroll
  for (int dt = 0; dt < 16; ++dt)
#pragma unroll
    for (int j = 0; j < 8; ++j) {
      size_t row = (size_t)(b * 1024 + n0 + wave * 16 + j + half * 8);
      o[row * MDIM + h * HDIM + dt * 16 + l16] = f2bf(accd[dt][j] / lrow[j]);
    }
}

// ---------------------------------------------------------------- router (E = 8, tiny)
__global__ void __launch_bounds__(256) router_kernel(
    const float* __restrict__ z, const float* __restrict__ noise,
    const float* __restrict__ Wg, const float* __restrict__ bg,
    const float* __restrict__ Wn, const float* __restrict__ bn,
    float* __restrict__ gates, float* __restrict__ Pout) {
  const int t = blockIdx.x;
  const float* zr = z + (size_t)t * MDIM;
  float lg[8] = {0}, nsv[8] = {0};
  for (int m = threadIdx.x; m < MDIM; m += 256) {
    float zv = zr[m];
    const float* wg = Wg + (size_t)m * 8;
    const float* wn = Wn + (size_t)m * 8;
#pragma unroll
    for (int e = 0; e < 8; ++e) { lg[e] += zv * wg[e]; nsv[e] += zv * wn[e]; }
  }
  __shared__ float red[256 * 16];
#pragma unroll
  for (int e = 0; e < 8; ++e) { red[threadIdx.x * 16 + e] = lg[e]; red[threadIdx.x * 16 + 8 + e] = nsv[e]; }
  __syncthreads();
  for (int off = 128; off > 0; off >>= 1) {
    if (threadIdx.x < off)
      for (int i = 0; i < 16; ++i) red[threadIdx.x * 16 + i] += red[(threadIdx.x + off) * 16 + i];
    __syncthreads();
  }
  if (threadIdx.x == 0) {
    float logit[8], ns[8], Hn[8];
    for (int e = 0; e < 8; ++e) {
      logit[e] = red[e] + bg[e];
      float raw = red[8 + e] + bn[e];
      ns[e] = raw > 20.f ? raw : log1pf(__expf(raw));          // softplus
      Hn[e] = logit[e] + noise[(size_t)t * 8 + e] * ns[e];
    }
    int i1 = 0, i2 = 0; float v1 = -1e30f, v2 = -1e30f, v3 = -1e30f;
    for (int e = 0; e < 8; ++e) {
      float hv = Hn[e];
      if (hv > v1)      { v3 = v2; v2 = v1; i2 = i1; v1 = hv; i1 = e; }
      else if (hv > v2) { v3 = v2; v2 = hv; i2 = e; }
      else if (hv > v3) { v3 = hv; }
    }
    float e2 = __expf(v2 - v1), ssum = 1.f + e2;
    for (int e = 0; e < 8; ++e) gates[(size_t)t * 8 + e] = 0.f;
    gates[(size_t)t * 8 + i1] = 1.f / ssum;
    gates[(size_t)t * 8 + i2] = e2 / ssum;
    for (int e = 0; e < 8; ++e) {
      float hv = Hn[e];
      float psi = hv > v2 ? v2 : (hv <= v3 ? v3 : hv);
      float xa = (logit[e] - psi) / ns[e];
      Pout[(size_t)t * 8 + e] = 0.5f * (1.f + erff(xa * 0.70710678f));
    }
  }
}

// ---------------------------------------------------------------- aux losses
__global__ void __launch_bounds__(256) loss_kernel(
    const float* __restrict__ gates, const float* __restrict__ P, float* __restrict__ out) {
  float gs[8] = {0}, ps[8] = {0};
  for (int t = threadIdx.x; t < T_TOK; t += 256)
#pragma unroll
    for (int e = 0; e < 8; ++e) { gs[e] += gates[(size_t)t * 8 + e]; ps[e] += P[(size_t)t * 8 + e]; }
  __shared__ float red[256 * 16];
#pragma unroll
  for (int e = 0; e < 8; ++e) { red[threadIdx.x * 16 + e] = gs[e]; red[threadIdx.x * 16 + 8 + e] = ps[e]; }
  __syncthreads();
  for (int off = 128; off > 0; off >>= 1) {
    if (threadIdx.x < off)
      for (int i = 0; i < 16; ++i) red[threadIdx.x * 16 + i] += red[(threadIdx.x + off) * 16 + i];
    __syncthreads();
  }
  if (threadIdx.x == 0) {
    float mug = 0, mup = 0;
    for (int e = 0; e < 8; ++e) { mug += red[e]; mup += red[8 + e]; }
    mug *= 0.125f; mup *= 0.125f;
    float vg = 0, vp = 0;
    for (int e = 0; e < 8; ++e) {
      float dg = red[e] - mug, dp = red[8 + e] - mup;
      vg += dg * dg; vp += dp * dp;
    }
    float cvg = sqrtf(vg * 0.125f) / (mug + 1e-6f);
    float cvp = sqrtf(vp * 0.125f) / (mup + 1e-6f);
    out[0] = 0.01f * cvg + 0.01f * cvp;
  }
}

__global__ void __launch_bounds__(256) f2bf_kernel(const float* __restrict__ in,
                                                   bf16_t* __restrict__ out, size_t n) {
  size_t i = (size_t)blockIdx.x * 256 + threadIdx.x;
  if (i < n) out[i] = f2bf(in[i]);
}

// ---------------------------------------------------------------- host
extern "C" void kernel_launch(void* const* d_in, const int* in_sizes, int n_in,
                              void* d_out, int out_size, void* d_ws, size_t ws_size,
                              hipStream_t stream) {
  const float* x     = (const float*)d_in[0];
  const float* noise = (const float*)d_in[1];
  const float* ln1_g = (const float*)d_in[2];
  const float* ln1_b = (const float*)d_in[3];
  const float* ln2_g = (const float*)d_in[4];
  const float* ln2_b = (const float*)d_in[5];
  const float* Wq = (const float*)d_in[6];  const float* bq = (const float*)d_in[7];
  const float* Wk = (const float*)d_in[8];  const float* bk = (const float*)d_in[9];
  const float* Wv = (const float*)d_in[10]; const float* bv = (const float*)d_in[11];
  const float* Wo = (const float*)d_in[12]; const float* bo = (const float*)d_in[13];
  const float* Wg = (const float*)d_in[14]; const float* bg = (const float*)d_in[15];
  const float* Wn = (const float*)d_in[16]; const float* bn = (const float*)d_in[17];
  const float* We = (const float*)d_in[18]; const float* be = (const float*)d_in[19];
  const float* Vw = (const float*)d_in[20]; const float* Vb = (const float*)d_in[21];
  const float* W2w = (const float*)d_in[22]; const float* W2b = (const float*)d_in[23];

  char* ws = (char*)d_ws;
  size_t off = 0;
  auto take = [&](size_t bytes) -> char* {
    char* p = ws + off; off += (bytes + 255) & ~(size_t)255; return p;
  };
  // ~420 MB scratch (single rotating slot for per-expert weights)
  bf16_t* wt_q  = (bf16_t*)take((size_t)MDIM * MDIM * 2);
  bf16_t* wt_k  = (bf16_t*)take((size_t)MDIM * KVDIM * 2);
  bf16_t* wt_v  = (bf16_t*)take((size_t)MDIM * KVDIM * 2);
  bf16_t* wt_o  = (bf16_t*)take((size_t)MDIM * MDIM * 2);
  bf16_t* wt_vw = (bf16_t*)take((size_t)MDIM * DHDIM * 2);
  bf16_t* wt_w2 = (bf16_t*)take((size_t)DHDIM * MDIM * 2);
  bf16_t* wt_e  = (bf16_t*)take((size_t)MDIM * DHDIM * 2);
  bf16_t* yb    = (bf16_t*)take((size_t)T_TOK * MDIM * 2);
  bf16_t* qb    = (bf16_t*)take((size_t)T_TOK * MDIM * 2);
  bf16_t* kbuf  = (bf16_t*)take((size_t)T_TOK * KVDIM * 2);
  bf16_t* vbuf  = (bf16_t*)take((size_t)T_TOK * KVDIM * 2);
  bf16_t* ab    = (bf16_t*)take((size_t)T_TOK * MDIM * 2);
  float*  x1    = (float*)take((size_t)T_TOK * MDIM * 4);
  bf16_t* zb    = (bf16_t*)take((size_t)T_TOK * MDIM * 2);
  float*  zf    = (float*)take((size_t)T_TOK * MDIM * 4);
  float*  xv    = (float*)take((size_t)T_TOK * DHDIM * 4);
  float*  accf  = (float*)take((size_t)T_TOK * DHDIM * 4);
  bf16_t* accb  = (bf16_t*)take((size_t)T_TOK * DHDIM * 2);
  float*  gates = (float*)take((size_t)T_TOK * NEXP * 4);
  float*  pp    = (float*)take((size_t)T_TOK * NEXP * 4);
  (void)ws_size; (void)in_sizes; (void)n_in; (void)out_size;

  float* outF = (float*)d_out;
  const dim3 B256(256), B128(128);

  // one-time weight transpose+convert to bf16 (N x K)
  transpose_convert_kernel<<<dim3(MDIM / 32, MDIM / 32), B256, 0, stream>>>(Wq, wt_q, MDIM, MDIM);
  transpose_convert_kernel<<<dim3(KVDIM / 32, MDIM / 32), B256, 0, stream>>>(Wk, wt_k, MDIM, KVDIM);
  transpose_convert_kernel<<<dim3(KVDIM / 32, MDIM / 32), B256, 0, stream>>>(Wv, wt_v, MDIM, KVDIM);
  transpose_convert_kernel<<<dim3(MDIM / 32, MDIM / 32), B256, 0, stream>>>(Wo, wt_o, MDIM, MDIM);
  transpose_convert_kernel<<<dim3(DHDIM / 32, MDIM / 32), B256, 0, stream>>>(Vw, wt_vw, MDIM, DHDIM);
  transpose_convert_kernel<<<dim3(MDIM / 32, DHDIM / 32), B256, 0, stream>>>(W2w, wt_w2, DHDIM, MDIM);

  // LN1 -> y (bf16)
  layernorm_kernel<<<T_TOK, B256, 0, stream>>>(x, ln1_g, ln1_b, yb, nullptr);

  // QKV projections (bf16 out + bias)
  gemm_kernel<1><<<dim3(MDIM / 128, T_TOK / 128), B256, 0, stream>>>(
      yb, MDIM, wt_q, MDIM, bq, MDIM, MDIM, nullptr, qb, nullptr, nullptr, 0, nullptr, nullptr);
  gemm_kernel<1><<<dim3(KVDIM / 128, T_TOK / 128), B256, 0, stream>>>(
      yb, MDIM, wt_k, MDIM, bk, MDIM, KVDIM, nullptr, kbuf, nullptr, nullptr, 0, nullptr, nullptr);
  gemm_kernel<1><<<dim3(KVDIM / 128, T_TOK / 128), B256, 0, stream>>>(
      yb, MDIM, wt_v, MDIM, bv, MDIM, KVDIM, nullptr, vbuf, nullptr, nullptr, 0, nullptr, nullptr);

  // flash attention
  attention_kernel<<<dim3(1024 / 64, NHEAD, 2), B128, 0, stream>>>(qb, kbuf, vbuf, ab);

  // x1 = x + attn @ Wo + bo
  gemm_kernel<2><<<dim3(MDIM / 128, T_TOK / 128), B256, 0, stream>>>(
      ab, MDIM, wt_o, MDIM, bo, MDIM, MDIM, x1, nullptr, x, nullptr, 0, nullptr, nullptr);

  // LN2 -> z (bf16 + f32)
  layernorm_kernel<<<T_TOK, B256, 0, stream>>>(x1, ln2_g, ln2_b, zb, zf);

  // router + aux-loss inputs
  router_kernel<<<T_TOK, B256, 0, stream>>>(zf, noise, Wg, bg, Wn, bn, gates, pp);

  // xV = z @ Vw + Vb (f32)
  gemm_kernel<0><<<dim3(DHDIM / 128, T_TOK / 128), B256, 0, stream>>>(
      zb, MDIM, wt_vw, MDIM, Vb, MDIM, DHDIM, xv, nullptr, nullptr, nullptr, 0, nullptr, nullptr);

  // MoE experts: acc += gate * silu(z @ We[e] + be[e]) * xV   (fused epilogue)
  hipMemsetAsync(accf, 0, (size_t)T_TOK * DHDIM * 4, stream);
  for (int e = 0; e < NEXP; ++e) {
    transpose_convert_kernel<<<dim3(DHDIM / 32, MDIM / 32), B256, 0, stream>>>(
        We + (size_t)e * MDIM * DHDIM, wt_e, MDIM, DHDIM);
    gemm_kernel<3><<<dim3(DHDIM / 128, T_TOK / 128), B256, 0, stream>>>(
        zb, MDIM, wt_e, MDIM, be + (size_t)e * DHDIM, MDIM, DHDIM,
        nullptr, nullptr, nullptr, gates, e, xv, accf);
  }
  f2bf_kernel<<<(T_TOK * (size_t)DHDIM + 255) / 256, B256, 0, stream>>>(
      accf, accb, (size_t)T_TOK * DHDIM);

  // out = x1 + acc @ W2w + W2b
  gemm_kernel<2><<<dim3(MDIM / 128, T_TOK / 128), B256, 0, stream>>>(
      accb, DHDIM, wt_w2, DHDIM, W2b, DHDIM, MDIM, outF, nullptr, x1, nullptr, 0, nullptr, nullptr);

  // aux loss scalar
  loss_kernel<<<1, B256, 0, stream>>>(gates, pp, outF + (size_t)T_TOK * MDIM);
}
